// NodeClsGraphSAGE_7533372637723
// MI455X (gfx1250) — matrix operand order repaired and checked by reading, not compiled, not run on hardware
//
#include <hip/hip_runtime.h>
#include <hip/hip_bf16.h>

// ---------------------------------------------------------------------------
// GraphSAGE (aggr='min') x3 + BatchNorm/ReLU + log_softmax for gfx1250.
// bf16 WMMA (v_wmma_f32_16x16x32_bf16) with fp32 accumulate for all GEMMs.
// A operands: row-major activations, loaded as contiguous b128 pairs (the ISA
// 16-bit A layout is two contiguous K-runs per lane). B operands: weights
// pre-swizzled once into the per-lane WMMA layout -> one 32B load per lane.
// Scatter-min via ordered-uint global atomics (exact -> deterministic).
// All streaming elementwise kernels use 128-bit accesses.
// ---------------------------------------------------------------------------

typedef __attribute__((ext_vector_type(16))) __bf16   v16bf;
typedef __attribute__((ext_vector_type(4)))  __bf16   v4bf;
typedef __attribute__((ext_vector_type(8)))  float    v8f;
typedef __attribute__((ext_vector_type(8)))  unsigned v8u;
typedef __attribute__((ext_vector_type(4)))  unsigned v4u;

#define NNODES 50000
#define NEDGES 800000
#define CH     64          // IN_C == H0 == H1 == 64
#define OUTC   16

// ---------------- elementwise helpers (4-wide) ----------------

__global__ void k_cast_bf16_x4(const float4* __restrict__ s, v4bf* __restrict__ d, int n4) {
    int i = blockIdx.x * blockDim.x + threadIdx.x;
    if (i >= n4) return;
    float4 v = s[i];
    v4bf o;
    o[0] = (__bf16)v.x; o[1] = (__bf16)v.y; o[2] = (__bf16)v.z; o[3] = (__bf16)v.w;
    d[i] = o;
}

__global__ void k_fill_u32_x4(uint4* __restrict__ p, unsigned v, int n4) {
    int i = blockIdx.x * blockDim.x + threadIdx.x;
    if (i < n4) p[i] = make_uint4(v, v, v, v);
}

// Swizzle a 64xN fp32 weight matrix into the WMMA B-operand lane layout (bf16):
//   out[(((nt*2+s)*32 + lane)*16 + p)] = W[(32s + 8h + (p&7) + 16*(p>>3))*N + nt*16 + ln]
__global__ void k_swizzle_w(const float* __restrict__ W, __bf16* __restrict__ Bs, int N) {
    int j = blockIdx.x * blockDim.x + threadIdx.x;
    if (j >= 64 * N) return;
    int p    = j & 15;
    int lane = (j >> 4) & 31;
    int s    = (j >> 9) & 1;
    int nt   = j >> 10;
    int h  = lane >> 4;
    int ln = lane & 15;
    int k   = 32 * s + 8 * h + (p & 7) + ((p >> 3) << 4);
    int col = nt * 16 + ln;
    Bs[j] = (__bf16)W[k * N + col];
}

// ---------------- scatter-min aggregation ----------------
// key encoding: monotone map float -> uint so unsigned min == float min.
// enc(+inf) = 0xFF800000 (init value); empty segments stay there -> 0 later.

__device__ __forceinline__ unsigned enc_f32(float v) {
    unsigned u = __float_as_uint(v);
    return (u & 0x80000000u) ? ~u : (u | 0x80000000u);
}

__device__ __forceinline__ float dec_u32(unsigned k) {
    unsigned u = (k & 0x80000000u) ? (k ^ 0x80000000u) : ~k;
    return (u == 0x7f800000u) ? 0.0f : __uint_as_float(u);   // empty -> 0
}

__global__ void k_scatter_min(const float* __restrict__ feat,
                              const int* __restrict__ src,
                              const int* __restrict__ dst,
                              unsigned* __restrict__ key, int nE) {
    long long tid = (long long)blockIdx.x * blockDim.x + threadIdx.x;
    int e = (int)(tid >> 4);          // 16 threads/edge, 4 channels/thread
    int c = (int)(tid & 15) * 4;
    if (e >= nE) return;
    int sn = src[e];
    int dn = dst[e];
    const float4 v = *(const float4*)(feat + (long long)sn * CH + c);
    unsigned* kp = key + (long long)dn * CH + c;
    atomicMin(kp + 0, enc_f32(v.x));
    atomicMin(kp + 1, enc_f32(v.y));
    atomicMin(kp + 2, enc_f32(v.z));
    atomicMin(kp + 3, enc_f32(v.w));
}

__global__ void k_finalize_agg_x4(const uint4* __restrict__ key,
                                  v4bf* __restrict__ aggb, int n4) {
    int i = blockIdx.x * blockDim.x + threadIdx.x;
    if (i >= n4) return;
    uint4 k = key[i];
    v4bf o;
    o[0] = (__bf16)dec_u32(k.x);
    o[1] = (__bf16)dec_u32(k.y);
    o[2] = (__bf16)dec_u32(k.z);
    o[3] = (__bf16)dec_u32(k.w);
    aggb[i] = o;
}

// ---------------- dual-GEMM via WMMA ----------------
// out[M x N] = A1[M x 64] @ B1[64 x N] + A2[M x 64] @ B2[64 x N] + bias[N]
// One wave per 16-row tile; covers all N tiles. K = 64 -> 2 k-steps of 32.
// B1/B2 are pre-swizzled (k_swizzle_w). A loads: two b128 per lane per k-step.

__device__ __forceinline__ v16bf load_a(const __bf16* __restrict__ base) {
    const v4u* p = (const v4u*)base;     // 16B chunks
    v4u lo = p[0];                       // K run 0..7   (within this lane's half)
    v4u hi = p[2];                       // K run 16..23 (+32B)
    v8u t = __builtin_shufflevector(lo, hi, 0, 1, 2, 3, 4, 5, 6, 7);
    return __builtin_bit_cast(v16bf, t);
}

template <int N>
__global__ __launch_bounds__(32)
void k_gemm_dual(const __bf16* __restrict__ A1, const __bf16* __restrict__ B1s,
                 const __bf16* __restrict__ A2, const __bf16* __restrict__ B2s,
                 const float* __restrict__ bias, float* __restrict__ out) {
    constexpr int NT = N / 16;
    const int m0   = blockIdx.x * 16;
    const int lane = threadIdx.x;
    const int h    = lane >> 4;      // half of wave
    const int ln   = lane & 15;
    const int row  = m0 + ln;

    v16bf a1[2], a2[2];
#pragma unroll
    for (int s = 0; s < 2; ++s) {
        a1[s] = load_a(A1 + (long long)row * 64 + 32 * s + 8 * h);
        a2[s] = load_a(A2 + (long long)row * 64 + 32 * s + 8 * h);
    }

#pragma unroll
    for (int nt = 0; nt < NT; ++nt) {
        const int col = nt * 16 + ln;
        float bv = bias[col];
        v8f acc;
#pragma unroll
        for (int j = 0; j < 8; ++j) acc[j] = bv;

#pragma unroll
        for (int s = 0; s < 2; ++s) {
            v16bf b1v = *(const v16bf*)(B1s + (long long)((nt * 2 + s) * 32 + lane) * 16);
            v16bf b2v = *(const v16bf*)(B2s + (long long)((nt * 2 + s) * 32 + lane) * 16);
            acc = __builtin_amdgcn_wmma_f32_16x16x32_bf16(
                false, a1[s], false, b1v, (short)0, acc, false, false);
            acc = __builtin_amdgcn_wmma_f32_16x16x32_bf16(
                false, a2[s], false, b2v, (short)0, acc, false, false);
        }

        // C/D layout: VGPR j -> M = m0 + j + 8h, N = col
#pragma unroll
        for (int j = 0; j < 8; ++j)
            out[(long long)(m0 + 8 * h + j) * N + col] = acc[j];
    }
}

// ---------------- BatchNorm (training stats, biased var) ----------------
// Deterministic two-stage reduction; stage-1 loads are channel-coalesced.

__global__ __launch_bounds__(256)
void k_bn_partial(const float* __restrict__ z, float* __restrict__ part, int M) {
    __shared__ float s1[256];
    __shared__ float s2[256];
    const int c    = threadIdx.x & 63;
    const int rsub = threadIdx.x >> 6;            // 0..3
    const long long r0 = (long long)blockIdx.x * 256;
    float a = 0.0f, b = 0.0f;
    for (int i = rsub; i < 256; i += 4) {
        long long r = r0 + i;
        if (r < M) {
            float v = z[r * CH + c];
            a += v;
            b += v * v;
        }
    }
    s1[threadIdx.x] = a;
    s2[threadIdx.x] = b;
    __syncthreads();
    if (rsub == 0) {   // fixed-order fold of the 4 partials -> deterministic
        float A = s1[c] + s1[c + 64] + s1[c + 128] + s1[c + 192];
        float B = s2[c] + s2[c + 64] + s2[c + 128] + s2[c + 192];
        part[(long long)blockIdx.x * 128 + c]      = A;
        part[(long long)blockIdx.x * 128 + 64 + c] = B;
    }
}

__global__ __launch_bounds__(64)
void k_bn_final(const float* __restrict__ part, float* __restrict__ stats,
                int nBlk, int M) {
    int c = threadIdx.x;               // 0..63
    float a = 0.0f, b = 0.0f;
    for (int k = 0; k < nBlk; ++k) {   // fixed order -> deterministic
        a += part[(long long)k * 128 + c];
        b += part[(long long)k * 128 + 64 + c];
    }
    float mu  = a / (float)M;
    float var = b / (float)M - mu * mu;
    stats[c]      = mu;
    stats[CH + c] = rsqrtf(var + 1e-5f);
}

__global__ void k_bn_relu_cast_x4(const float4* __restrict__ z,
                                  const float* __restrict__ stats,
                                  const float* __restrict__ g,
                                  const float* __restrict__ be,
                                  float4* __restrict__ hf, v4bf* __restrict__ hb, int n4) {
    int i = blockIdx.x * blockDim.x + threadIdx.x;
    if (i >= n4) return;
    int c = (i * 4) & (CH - 1);        // 4 consecutive channels
    float4 zv = z[i];
    float4 mu = *(const float4*)(stats + c);
    float4 rs = *(const float4*)(stats + CH + c);
    float4 gg = *(const float4*)(g + c);
    float4 bb = *(const float4*)(be + c);
    float4 o;
    o.x = fmaxf(gg.x * (zv.x - mu.x) * rs.x + bb.x, 0.0f);
    o.y = fmaxf(gg.y * (zv.y - mu.y) * rs.y + bb.y, 0.0f);
    o.z = fmaxf(gg.z * (zv.z - mu.z) * rs.z + bb.z, 0.0f);
    o.w = fmaxf(gg.w * (zv.w - mu.w) * rs.w + bb.w, 0.0f);
    hf[i] = o;
    v4bf ob;
    ob[0] = (__bf16)o.x; ob[1] = (__bf16)o.y; ob[2] = (__bf16)o.z; ob[3] = (__bf16)o.w;
    hb[i] = ob;
}

// ---------------- log_softmax over 16 channels, in place ----------------

__global__ void k_logsoftmax16(float4* __restrict__ out, int M) {
    int r = blockIdx.x * blockDim.x + threadIdx.x;
    if (r >= M) return;
    float4 q[4];
#pragma unroll
    for (int i = 0; i < 4; ++i) q[i] = out[(long long)r * 4 + i];
    float* v = (float*)q;
    float mx = -__builtin_inff();
#pragma unroll
    for (int c = 0; c < OUTC; ++c) mx = fmaxf(mx, v[c]);
    float s = 0.0f;
#pragma unroll
    for (int c = 0; c < OUTC; ++c) s += __expf(v[c] - mx);
    float off = mx + __logf(s);
#pragma unroll
    for (int c = 0; c < OUTC; ++c) v[c] -= off;
#pragma unroll
    for (int i = 0; i < 4; ++i) out[(long long)r * 4 + i] = q[i];
}

// ---------------- host orchestration ----------------

extern "C" void kernel_launch(void* const* d_in, const int* in_sizes, int n_in,
                              void* d_out, int out_size, void* d_ws, size_t ws_size,
                              hipStream_t stream) {
    const float* x   = (const float*)d_in[0];
    const int*   ei  = (const int*)d_in[1];
    const float* W1l = (const float*)d_in[2];
    const float* b1  = (const float*)d_in[3];
    const float* W1r = (const float*)d_in[4];
    const float* g1  = (const float*)d_in[5];
    const float* be1 = (const float*)d_in[6];
    const float* W2l = (const float*)d_in[7];
    const float* b2  = (const float*)d_in[8];
    const float* W2r = (const float*)d_in[9];
    const float* g2  = (const float*)d_in[10];
    const float* be2 = (const float*)d_in[11];
    const float* W3l = (const float*)d_in[12];
    const float* b3  = (const float*)d_in[13];
    const float* W3r = (const float*)d_in[14];

    const int* src = ei;
    const int* dst = ei + NEDGES;

    const int nFeat  = NNODES * CH;               // 3,200,000
    const int nF4    = nFeat / 4;                 //   800,000
    const int nBnBlk = (NNODES + 255) / 256;      // 196

    // workspace layout (256B aligned)
    char*  ws  = (char*)d_ws;
    size_t off = 0;
    auto carve = [&](size_t bytes) -> char* {
        char* p = ws + off;
        off = (off + bytes + 255) & ~(size_t)255;
        return p;
    };
    unsigned* aggkey = (unsigned*)carve((size_t)nFeat * 4);
    __bf16*   aggb   = (__bf16*)  carve((size_t)nFeat * 2);
    __bf16*   xb     = (__bf16*)  carve((size_t)nFeat * 2);   // bf16 of layer input
    float*    hf     = (float*)   carve((size_t)nFeat * 4);   // fp32 hidden
    float*    z      = (float*)   carve((size_t)nFeat * 4);   // pre-BN GEMM out
    float*    stats  = (float*)   carve(2 * CH * 4);
    float*    part   = (float*)   carve((size_t)nBnBlk * 128 * 4);
    __bf16*   w1lb   = (__bf16*)  carve(64 * 64 * 2);
    __bf16*   w1rb   = (__bf16*)  carve(64 * 64 * 2);
    __bf16*   w2lb   = (__bf16*)  carve(64 * 64 * 2);
    __bf16*   w2rb   = (__bf16*)  carve(64 * 64 * 2);
    __bf16*   w3lb   = (__bf16*)  carve(64 * OUTC * 2);
    __bf16*   w3rb   = (__bf16*)  carve(64 * OUTC * 2);
    (void)ws_size; (void)n_in; (void)in_sizes; (void)out_size;

    const int T = 256;
    const int f4Blocks   = (nF4 + T - 1) / T;
    const int edgeBlocks = (int)(((long long)NEDGES * 16 + T - 1) / T);
    const int mTiles = NNODES / 16;               // 3125, exact

    // input cast + weight swizzles (weights are tiny: one-shot)
    k_cast_bf16_x4<<<f4Blocks, T, 0, stream>>>((const float4*)x, (v4bf*)xb, nF4);
    k_swizzle_w<<<(64 * 64 + T - 1) / T, T, 0, stream>>>(W1l, w1lb, 64);
    k_swizzle_w<<<(64 * 64 + T - 1) / T, T, 0, stream>>>(W1r, w1rb, 64);
    k_swizzle_w<<<(64 * 64 + T - 1) / T, T, 0, stream>>>(W2l, w2lb, 64);
    k_swizzle_w<<<(64 * 64 + T - 1) / T, T, 0, stream>>>(W2r, w2rb, 64);
    k_swizzle_w<<<(64 * OUTC + T - 1) / T, T, 0, stream>>>(W3l, w3lb, OUTC);
    k_swizzle_w<<<(64 * OUTC + T - 1) / T, T, 0, stream>>>(W3r, w3rb, OUTC);

    // ---- layer 1 ----
    k_fill_u32_x4<<<f4Blocks, T, 0, stream>>>((uint4*)aggkey, 0xFF800000u, nF4);
    k_scatter_min<<<edgeBlocks, T, 0, stream>>>(x, src, dst, aggkey, NEDGES);
    k_finalize_agg_x4<<<f4Blocks, T, 0, stream>>>((const uint4*)aggkey, (v4bf*)aggb, nF4);
    k_gemm_dual<64><<<mTiles, 32, 0, stream>>>(aggb, w1lb, xb, w1rb, b1, z);
    k_bn_partial<<<nBnBlk, 256, 0, stream>>>(z, part, NNODES);
    k_bn_final<<<1, 64, 0, stream>>>(part, stats, nBnBlk, NNODES);
    k_bn_relu_cast_x4<<<f4Blocks, T, 0, stream>>>((const float4*)z, stats, g1, be1,
                                                  (float4*)hf, (v4bf*)xb, nF4);

    // ---- layer 2 (xb now holds bf16 of h1) ----
    k_fill_u32_x4<<<f4Blocks, T, 0, stream>>>((uint4*)aggkey, 0xFF800000u, nF4);
    k_scatter_min<<<edgeBlocks, T, 0, stream>>>(hf, src, dst, aggkey, NEDGES);
    k_finalize_agg_x4<<<f4Blocks, T, 0, stream>>>((const uint4*)aggkey, (v4bf*)aggb, nF4);
    k_gemm_dual<64><<<mTiles, 32, 0, stream>>>(aggb, w2lb, xb, w2rb, b2, z);
    k_bn_partial<<<nBnBlk, 256, 0, stream>>>(z, part, NNODES);
    k_bn_final<<<1, 64, 0, stream>>>(part, stats, nBnBlk, NNODES);
    k_bn_relu_cast_x4<<<f4Blocks, T, 0, stream>>>((const float4*)z, stats, g2, be2,
                                                  (float4*)hf, (v4bf*)xb, nF4);

    // ---- layer 3 -> d_out, then in-place log_softmax ----
    k_fill_u32_x4<<<f4Blocks, T, 0, stream>>>((uint4*)aggkey, 0xFF800000u, nF4);
    k_scatter_min<<<edgeBlocks, T, 0, stream>>>(hf, src, dst, aggkey, NEDGES);
    k_finalize_agg_x4<<<f4Blocks, T, 0, stream>>>((const uint4*)aggkey, (v4bf*)aggb, nF4);
    k_gemm_dual<OUTC><<<mTiles, 32, 0, stream>>>(aggb, w3lb, xb, w3rb, b3, (float*)d_out);
    k_logsoftmax16<<<(NNODES + T - 1) / T, T, 0, stream>>>((float4*)d_out, NNODES);
}
